// MotionTransformerVAE_55473797595857
// MI455X (gfx1250) — matrix operand order repaired and checked by reading, not compiled, not run on hardware
//
#include <hip/hip_runtime.h>
#include <hip/hip_bf16.h>

typedef __attribute__((ext_vector_type(16))) _Float16 v16h;
typedef __attribute__((ext_vector_type(8)))  _Float16 v8h;
typedef __attribute__((ext_vector_type(8)))  float    v8f;

#define BATCH 64
#define SEQ   200
#define DM    1024
#define NH    16
#define DKH   64
#define SP    224            // keys padded to 7*32 for WMMA K-steps
#define MTOT  (BATCH*SEQ)    // 12800

union FragU { v16h v; v8h h[2]; };

__device__ __forceinline__ v16h ldg_frag(const _Float16* p0, const _Float16* p1) {
  FragU f;
  f.h[0] = *(const v8h*)p0;
  f.h[1] = *(const v8h*)p1;
  return f.v;
}

__global__ void k_cvt(const float* __restrict__ in, _Float16* __restrict__ out, int n) {
  int i = blockIdx.x * 256 + threadIdx.x;
  if (i < n) out[i] = (_Float16)in[i];
}

// Wt[k*1024 + n] = (f16) W[n*1024 + k]
__global__ void k_transpose(const float* __restrict__ w, _Float16* __restrict__ wt) {
  int i = blockIdx.x * 256 + threadIdx.x;   // i = n*1024 + k
  int n = i >> 10, k = i & 1023;
  wt[(size_t)k * DM + n] = (_Float16)w[i];
}

__global__ void k_zero(_Float16* __restrict__ p, int n) {
  int i = blockIdx.x * 256 + threadIdx.x;
  if (i < n) p[i] = (_Float16)0.0f;
}

// C = A[12800x1024] * Bt[1024x1024] + bias ; wave computes 64x64 tile
// mode 0: f16 flat store (Q, V)   mode 1: f16 store into Kt[b,h,dk,SP]   mode 2: f32 flat store (final out)
__global__ void __launch_bounds__(256) k_gemm(const _Float16* __restrict__ A,
                                              const _Float16* __restrict__ Bt,
                                              const float* __restrict__ bias,
                                              void* __restrict__ outp, int mode) {
  int wave = blockIdx.x * 8 + (threadIdx.x >> 5);
  int lane = threadIdx.x & 31;
  int mw = wave >> 4;          // 0..199  (M tile of 64)
  int nw = wave & 15;          // 0..15   (N tile of 64)
  int g  = lane >> 4;
  int ln = lane & 15;

  v8f c[4][4];
#pragma unroll
  for (int i = 0; i < 4; i++)
#pragma unroll
    for (int j = 0; j < 4; j++) c[i][j] = v8f{};

  for (int k0 = 0; k0 < DM; k0 += 32) {
    v16h a[4];
#pragma unroll
    for (int mt = 0; mt < 4; mt++) {
      const _Float16* pa = A + (size_t)(mw * 64 + mt * 16 + ln) * DM + k0 + 8 * g;
      a[mt] = ldg_frag(pa, pa + 16);
    }
#pragma unroll
    for (int nt = 0; nt < 4; nt++) {
      v16h bfr;
#pragma unroll
      for (int s = 0; s < 16; s++)
        bfr[s] = Bt[(size_t)(k0 + 8 * g + (s & 7) + ((s >> 3) << 4)) * DM + nw * 64 + nt * 16 + ln];
#pragma unroll
      for (int mt = 0; mt < 4; mt++)
        c[mt][nt] = __builtin_amdgcn_wmma_f32_16x16x32_f16(
            false, a[mt], false, bfr, (short)0, c[mt][nt], false, false);
    }
  }

#pragma unroll
  for (int nt = 0; nt < 4; nt++) {
    int col = nw * 64 + nt * 16 + ln;
    float bv = bias[col];
#pragma unroll
    for (int mt = 0; mt < 4; mt++) {
#pragma unroll
      for (int r = 0; r < 8; r++) {
        int row = mw * 64 + mt * 16 + r + 8 * g;
        float v = c[mt][nt][r] + bv;
        if (mode == 0) {
          ((_Float16*)outp)[(size_t)row * DM + col] = (_Float16)v;
        } else if (mode == 1) {
          int b_ = row / SEQ, s = row % SEQ;
          int h = col >> 6, dk = col & 63;
          ((_Float16*)outp)[((size_t)((b_ * NH + h) * DKH + dk)) * SP + s] = (_Float16)v;
        } else {
          ((float*)outp)[(size_t)row * DM + col] = v;
        }
      }
    }
  }
}

// one wave per (b,h,qtile16): scores -> softmax -> attn out + ctx
__global__ void __launch_bounds__(32) k_attn(const _Float16* __restrict__ Qh,
                                             const _Float16* __restrict__ Kt,
                                             const _Float16* __restrict__ Vh,
                                             _Float16* __restrict__ ctxh,
                                             float* __restrict__ attnO) {
  __shared__ __align__(16) _Float16 sP[16 * SP];
  int qt = blockIdx.x % 13;
  int bh = blockIdx.x / 13;
  int b  = bh >> 4, h = bh & 15;
  int lane = threadIdx.x & 31;
  int g = lane >> 4, ln = lane & 15;

  // Q A-fragments, dk split into two K=32 steps (rows >=200 clamped; discarded later)
  v16h a[2];
  {
    int qs = qt * 16 + ln; if (qs > SEQ - 1) qs = SEQ - 1;
    const _Float16* pq = Qh + (size_t)(b * SEQ + qs) * DM + h * DKH + 8 * g;
    a[0] = ldg_frag(pq, pq + 16);
    a[1] = ldg_frag(pq + 32, pq + 48);
  }

  v8f c[14];
  const _Float16* ktb = Kt + (size_t)bh * DKH * SP;
#pragma unroll
  for (int nt = 0; nt < 14; nt++) {
    v8f acc = v8f{};
#pragma unroll
    for (int ks = 0; ks < 2; ks++) {
      v16h bfr;
#pragma unroll
      for (int s = 0; s < 16; s++)
        bfr[s] = ktb[(size_t)(ks * 32 + 8 * g + (s & 7) + ((s >> 3) << 4)) * SP + nt * 16 + ln];
      acc = __builtin_amdgcn_wmma_f32_16x16x32_f16(
          false, a[ks], false, bfr, (short)0, acc, false, false);
    }
    c[nt] = acc;
  }

  // scale + temporal bias + key mask
  float rmax[8];
#pragma unroll
  for (int r = 0; r < 8; r++) rmax[r] = -3.0e38f;
#pragma unroll
  for (int nt = 0; nt < 14; nt++) {
    int j = nt * 16 + ln;
#pragma unroll
    for (int r = 0; r < 8; r++) {
      int i = qt * 16 + r + 8 * g;
      float v = c[nt][r] * 0.125f - 0.0005f * fabsf((float)(i - j));
      if (j >= SEQ) v = -3.0e38f;
      c[nt][r] = v;
      rmax[r] = fmaxf(rmax[r], v);
    }
  }
#pragma unroll
  for (int r = 0; r < 8; r++) {
    float m = rmax[r];
    m = fmaxf(m, __shfl_xor(m, 1));
    m = fmaxf(m, __shfl_xor(m, 2));
    m = fmaxf(m, __shfl_xor(m, 4));
    m = fmaxf(m, __shfl_xor(m, 8));
    rmax[r] = m;
  }
  float rsum[8];
#pragma unroll
  for (int r = 0; r < 8; r++) rsum[r] = 0.0f;
#pragma unroll
  for (int nt = 0; nt < 14; nt++)
#pragma unroll
    for (int r = 0; r < 8; r++) {
      float p = __expf(c[nt][r] - rmax[r]);
      c[nt][r] = p;
      rsum[r] += p;
    }
#pragma unroll
  for (int r = 0; r < 8; r++) {
    float s = rsum[r];
    s += __shfl_xor(s, 1);
    s += __shfl_xor(s, 2);
    s += __shfl_xor(s, 4);
    s += __shfl_xor(s, 8);
    rsum[r] = 1.0f / s;
  }

  // normalize, emit fp32 attn, stage f16 probs in LDS
  float* attnBase = attnO + (size_t)bh * SEQ * SEQ;
#pragma unroll
  for (int nt = 0; nt < 14; nt++) {
    int j = nt * 16 + ln;
#pragma unroll
    for (int r = 0; r < 8; r++) {
      int i = qt * 16 + r + 8 * g;
      float p = c[nt][r] * rsum[r];
      sP[(r + 8 * g) * SP + j] = (_Float16)p;
      if (i < SEQ && j < SEQ) attnBase[(size_t)i * SEQ + j] = p;
    }
  }
  __syncthreads();

  // ctx = attn(16x224) @ V(224x64)
  const _Float16* vb = Vh + (size_t)(b * SEQ) * DM + h * DKH;
#pragma unroll
  for (int nt2 = 0; nt2 < 4; nt2++) {
    v8f acc = v8f{};
#pragma unroll
    for (int kk = 0; kk < 7; kk++) {
      const _Float16* pa = &sP[ln * SP + kk * 32 + 8 * g];
      FragU af; af.h[0] = *(const v8h*)pa; af.h[1] = *(const v8h*)(pa + 16);
      v16h bf;
#pragma unroll
      for (int s = 0; s < 16; s++) {
        int ks = kk * 32 + 8 * g + (s & 7) + ((s >> 3) << 4); if (ks > SEQ - 1) ks = SEQ - 1;  // padded keys: attn==0
        bf[s] = vb[(size_t)ks * DM + nt2 * 16 + ln];
      }
      acc = __builtin_amdgcn_wmma_f32_16x16x32_f16(
          false, af.v, false, bf, (short)0, acc, false, false);
    }
#pragma unroll
    for (int r = 0; r < 8; r++) {
      int i = qt * 16 + r + 8 * g;
      if (i < SEQ)
        ctxh[(size_t)(b * SEQ + i) * DM + h * DKH + nt2 * 16 + ln] = (_Float16)acc[r];
    }
  }
}

extern "C" void kernel_launch(void* const* d_in, const int* in_sizes, int n_in,
                              void* d_out, int out_size, void* d_ws, size_t ws_size,
                              hipStream_t stream) {
  const float* x  = (const float*)d_in[0];
  const float* wq = (const float*)d_in[1];
  const float* bq = (const float*)d_in[2];
  const float* wk = (const float*)d_in[3];
  const float* bk = (const float*)d_in[4];
  const float* wv = (const float*)d_in[5];
  const float* bv = (const float*)d_in[6];
  const float* wo = (const float*)d_in[7];
  const float* bo = (const float*)d_in[8];

  _Float16* ws = (_Float16*)d_ws;
  const size_t NX = (size_t)MTOT * DM;          // 13107200
  _Float16* Xh  = ws;
  _Float16* Qh  = ws + NX;
  _Float16* Vh  = ws + 2 * NX;
  _Float16* Cx  = ws + 3 * NX;
  _Float16* WqT = ws + 4 * NX;
  _Float16* WkT = WqT + (size_t)DM * DM;
  _Float16* WvT = WkT + (size_t)DM * DM;
  _Float16* WoT = WvT + (size_t)DM * DM;
  _Float16* Kt  = WoT + (size_t)DM * DM;        // [64*16, 64, 224]
  const int NKT = BATCH * NH * DKH * SP;        // 14680064

  float* outp  = (float*)d_out;
  float* attnp = outp + NX;                     // attn follows out

  k_cvt<<<(int)((NX + 255) / 256), 256, 0, stream>>>(x, Xh, (int)NX);
  k_transpose<<<4096, 256, 0, stream>>>(wq, WqT);
  k_transpose<<<4096, 256, 0, stream>>>(wk, WkT);
  k_transpose<<<4096, 256, 0, stream>>>(wv, WvT);
  k_transpose<<<4096, 256, 0, stream>>>(wo, WoT);
  k_zero<<<(NKT + 255) / 256, 256, 0, stream>>>(Kt, NKT);

  k_gemm<<<400, 256, 0, stream>>>(Xh, WqT, bq, Qh, 0);
  k_gemm<<<400, 256, 0, stream>>>(Xh, WkT, bk, Kt, 1);
  k_gemm<<<400, 256, 0, stream>>>(Xh, WvT, bv, Vh, 0);

  k_attn<<<BATCH * NH * 13, 32, 0, stream>>>(Qh, Kt, Vh, Cx, attnp);

  k_gemm<<<400, 256, 0, stream>>>(Cx, WoT, bo, outp, 2);
}
